// DepthNet_53231824667085
// MI455X (gfx1250) — matrix operand order repaired and checked by reading, not compiled, not run on hardware
//
#include <hip/hip_runtime.h>

// DepthNet (MVSNet cost-volume regularization) for gfx1250 / MI455X.
// Fused: homography warp + bilinear sample + 5-view variance -> LDS (f16)
//        -> per-tap GEMM via v_wmma_*_16x16x32_f16 -> 27-tap shift-add conv
//        -> cost volume -> in-place softmax over depth.
//
// Round 2 changes:
//  * WMMA group loop is scalar-controlled (readfirstlane) so EXEC stays all-1s
//    across every v_wmma (ISA requirement), no exec-mask loop.
//  * P stored transposed [tap][pos]: each lane's 8 D rows are 8 consecutive
//    positions -> single ds_store_b128 per half-matrix instead of 16x b16.
//  * f16-accumulator WMMA (v_wmma_f16_16x16x32_f16) when the builtin exists:
//    D is already packed f16, no cvt chain. Fallback: confirmed f32 WMMA+pack.

#define Bn 1
#define Cn 32
#define Hn 128
#define Wn 160
#define Dn 64
#define HWn (Hn * Wn)

// Output tile per workgroup and halo-padded tile
#define TD 2
#define TH 4
#define TW 16
#define PD (TD + 2)            // 4
#define PH (TH + 2)            // 6
#define PW (TW + 2)            // 18
#define NPOS (PD * PH * PW)    // 432 = 27 * 16
#define NGRP (NPOS / 16)       // 27 WMMA row-groups

typedef _Float16 v16h __attribute__((ext_vector_type(16)));
typedef _Float16 h8   __attribute__((ext_vector_type(8)));
typedef float    v8f  __attribute__((ext_vector_type(8)));

// ---------------------------------------------------------------------------
// Setup: fuse projections, invert ref, store per-view rot(9)+trans(3) to ws.
// ---------------------------------------------------------------------------
__device__ void fuse4(const float* p, float* out) {
  const float* E = p;        // p[v][0]
  const float* K = p + 16;   // p[v][1]
  for (int i = 0; i < 4; ++i)
    for (int j = 0; j < 4; ++j) out[i * 4 + j] = E[i * 4 + j];
  for (int i = 0; i < 3; ++i)
    for (int j = 0; j < 4; ++j) {
      float s = 0.f;
      for (int k = 0; k < 3; ++k) s += K[i * 4 + k] * E[k * 4 + j];
      out[i * 4 + j] = s;
    }
}

__device__ void inv4(const float* a_in, float* out) {
  float a[16], id[16];
  for (int i = 0; i < 16; ++i) { a[i] = a_in[i]; id[i] = 0.f; }
  id[0] = id[5] = id[10] = id[15] = 1.f;
  for (int col = 0; col < 4; ++col) {
    int piv = col;
    float best = fabsf(a[col * 4 + col]);
    for (int r = col + 1; r < 4; ++r) {
      float v = fabsf(a[r * 4 + col]);
      if (v > best) { best = v; piv = r; }
    }
    if (piv != col)
      for (int j = 0; j < 4; ++j) {
        float t = a[col * 4 + j]; a[col * 4 + j] = a[piv * 4 + j]; a[piv * 4 + j] = t;
        t = id[col * 4 + j]; id[col * 4 + j] = id[piv * 4 + j]; id[piv * 4 + j] = t;
      }
    float d = 1.0f / a[col * 4 + col];
    for (int j = 0; j < 4; ++j) { a[col * 4 + j] *= d; id[col * 4 + j] *= d; }
    for (int r = 0; r < 4; ++r)
      if (r != col) {
        float f = a[r * 4 + col];
        for (int j = 0; j < 4; ++j) {
          a[r * 4 + j] -= f * a[col * 4 + j];
          id[r * 4 + j] -= f * id[col * 4 + j];
        }
      }
  }
  for (int i = 0; i < 16; ++i) out[i] = id[i];
}

__global__ void depthnet_setup(const float* __restrict__ proj,
                               float* __restrict__ rt) {
  float ref[16], refInv[16];
  fuse4(proj, ref);
  inv4(ref, refInv);
  for (int v = 1; v < 5; ++v) {
    float src[16], M[16];
    fuse4(proj + v * 32, src);
    for (int i = 0; i < 4; ++i)
      for (int j = 0; j < 4; ++j) {
        float s = 0.f;
        for (int k = 0; k < 4; ++k) s += src[i * 4 + k] * refInv[k * 4 + j];
        M[i * 4 + j] = s;
      }
    float* o = rt + (v - 1) * 12;
    o[0] = M[0];  o[1] = M[1];  o[2] = M[2];
    o[3] = M[4];  o[4] = M[5];  o[5] = M[6];
    o[6] = M[8];  o[7] = M[9];  o[8] = M[10];
    o[9] = M[3];  o[10] = M[7]; o[11] = M[11];
  }
}

// ---------------------------------------------------------------------------
// Fused warp + variance + WMMA per-tap GEMM + 27-tap shift-add conv.
// ---------------------------------------------------------------------------
__launch_bounds__(256, 1)
__global__ void depthnet_costvol(const float* __restrict__ f0,
                                 const float* __restrict__ f1,
                                 const float* __restrict__ f2,
                                 const float* __restrict__ f3,
                                 const float* __restrict__ f4,
                                 const float* __restrict__ rt,
                                 const float* __restrict__ depth,
                                 const float* __restrict__ conv_w,
                                 const float* __restrict__ conv_b,
                                 float* __restrict__ out) {
  __shared__ __attribute__((aligned(16))) _Float16 var_lds[NPOS * 32]; // [pos][ch]
  __shared__ __attribute__((aligned(16))) _Float16 p_lds[32 * NPOS];   // [tap][pos]
  __shared__ __attribute__((aligned(16))) _Float16 wt_lds[32 * 32];    // [ch][tap]

  const int tid = threadIdx.x;
  const int bid = blockIdx.x;
  const int tw = bid % 10, th = (bid / 10) % 32, tdi = bid / 320;
  const int w0 = tw * TW, h0 = th * TH, d0 = tdi * TD;

  // Warm L2/WGP$ for the weights (global_prefetch_b8 path).
  if (tid < 27) __builtin_prefetch(conv_w + tid * 32, 0, 0);

  // --- weights into LDS (zero-padded taps) ---
  for (int i = tid; i < 32 * 32; i += 256) wt_lds[i] = (_Float16)0.f;
  __syncthreads();
  for (int i = tid; i < 32 * 27; i += 256) {
    int c = i / 27, tap = i % 27;
    wt_lds[c * 32 + tap] = (_Float16)conv_w[i];
  }

  const float* fsrc[4] = {f1, f2, f3, f4};

  // --- Phase 1: per-halo-position, per-8-channel-chunk variance ---
  for (int item = tid; item < NPOS * 4; item += 256) {
    const int pos = item >> 2;
    const int chunk = item & 3;
    const int hw = pos % PW;
    const int hh = (pos / PW) % PH;
    const int hd = pos / (PW * PH);
    const int dg = d0 - 1 + hd;
    const int hg = h0 - 1 + hh;
    const int wg = w0 - 1 + hw;
    _Float16* dst = &var_lds[pos * 32 + chunk * 8];
    if (dg < 0 || dg >= Dn || hg < 0 || hg >= Hn || wg < 0 || wg >= Wn) {
#pragma unroll
      for (int c = 0; c < 8; ++c) dst[c] = (_Float16)0.f;
      continue;
    }
    const float dv = depth[dg];
    const float X = (float)wg, Y = (float)hg;
    const int base = hg * Wn + wg;
    float s[8], q[8];
#pragma unroll
    for (int c = 0; c < 8; ++c) {
      float a = f0[(chunk * 8 + c) * HWn + base];
      s[c] = a; q[c] = a * a;
    }
#pragma unroll
    for (int v = 0; v < 4; ++v) {
      const float* R = rt + v * 12;
      float zx = (R[0] * X + R[1] * Y + R[2]) * dv + R[9];
      float zy = (R[3] * X + R[4] * Y + R[5]) * dv + R[10];
      float zz = (R[6] * X + R[7] * Y + R[8]) * dv + R[11];
      float rz = 1.0f / zz;
      float px = zx * rz, py = zy * rz;
      float x0 = floorf(px), y0 = floorf(py);
      float fx = px - x0, fy = py - y0;
      bool vx0 = (x0 >= 0.f)  && (x0 <= (float)(Wn - 1));
      bool vx1 = (x0 >= -1.f) && (x0 <= (float)(Wn - 2));
      bool vy0 = (y0 >= 0.f)  && (y0 <= (float)(Hn - 1));
      bool vy1 = (y0 >= -1.f) && (y0 <= (float)(Hn - 2));
      float cx0 = fminf(fmaxf(x0, 0.f), (float)(Wn - 1));
      float cx1 = fminf(fmaxf(x0 + 1.f, 0.f), (float)(Wn - 1));
      float cy0 = fminf(fmaxf(y0, 0.f), (float)(Hn - 1));
      float cy1 = fminf(fmaxf(y0 + 1.f, 0.f), (float)(Hn - 1));
      int i00 = (int)cy0 * Wn + (int)cx0;
      int i01 = (int)cy0 * Wn + (int)cx1;
      int i10 = (int)cy1 * Wn + (int)cx0;
      int i11 = (int)cy1 * Wn + (int)cx1;
      float w00 = (1.f - fx) * (1.f - fy) * ((vx0 && vy0) ? 1.f : 0.f);
      float w01 = fx * (1.f - fy) * ((vx1 && vy0) ? 1.f : 0.f);
      float w10 = (1.f - fx) * fy * ((vx0 && vy1) ? 1.f : 0.f);
      float w11 = fx * fy * ((vx1 && vy1) ? 1.f : 0.f);
      const float* fv = fsrc[v];
#pragma unroll
      for (int c = 0; c < 8; ++c) {
        const float* fc = fv + (chunk * 8 + c) * HWn;
        float g = w00 * fc[i00] + w01 * fc[i01] + w10 * fc[i10] + w11 * fc[i11];
        s[c] += g; q[c] += g * g;
      }
    }
#pragma unroll
    for (int c = 0; c < 8; ++c) {
      float m = s[c] * 0.2f;
      dst[c] = (_Float16)(q[c] * 0.2f - m * m);
    }
  }
  __syncthreads();

  // --- Phase 2: P[tap][pos] = Var[pos, :] x Wt[:, tap] via WMMA f16 ---
  {
    // Scalar wave id: keeps the group loop in SALU so EXEC is all-1s for WMMA.
    const int wave = __builtin_amdgcn_readfirstlane(tid) >> 5;
    const int lane = tid & 31;
    const int nn = lane & 15;
    const bool hi = lane >= 16;
    // B: K(=channel) x N(=tap).  lane<16: K 0..7,16..23 of col nn; lane>=16: K 8..15,24..31.
    v16h b0, b1;
#pragma unroll
    for (int e = 0; e < 16; ++e) {
      int k = (hi ? 8 : 0) + (e < 8 ? e : 8 + e);
      b0[e] = wt_lds[k * 32 + nn];
      b1[e] = wt_lds[k * 32 + 16 + nn];
    }
    for (int g = wave; g < NGRP; g += 8) {
      const int prow = g * 16 + nn;            // A row M = nn
      const int cb = hi ? 8 : 0;               // lane>=16 holds K 8..15 / 24..31
      h8 lo = *(const h8*)(&var_lds[prow * 32 + cb]);
      h8 hp = *(const h8*)(&var_lds[prow * 32 + cb + 16]);
      v16h a;
#pragma unroll
      for (int e = 0; e < 8; ++e) { a[e] = lo[e]; a[8 + e] = hp[e]; }
      // Each lane's 8 D rows are positions g*16 + (hi?8:0) + 0..7 -> one b128.
      const int dst0 = nn * NPOS + g * 16 + (hi ? 8 : 0);        // taps 0..15
      const int dst1 = (16 + nn) * NPOS + g * 16 + (hi ? 8 : 0); // taps 16..31
#if __has_builtin(__builtin_amdgcn_wmma_f16_16x16x32_f16)
      h8 d0 = {}; h8 d1 = {};
      d0 = __builtin_amdgcn_wmma_f16_16x16x32_f16(false, a, false, b0,
                                                  (short)0, d0, false, false);
      d1 = __builtin_amdgcn_wmma_f16_16x16x32_f16(false, a, false, b1,
                                                  (short)0, d1, false, false);
      *(h8*)(&p_lds[dst0]) = d0;
      *(h8*)(&p_lds[dst1]) = d1;
#else
      v8f c0 = {}; v8f c1 = {};
      c0 = __builtin_amdgcn_wmma_f32_16x16x32_f16(false, a, false, b0,
                                                  (short)0, c0, false, false);
      c1 = __builtin_amdgcn_wmma_f32_16x16x32_f16(false, a, false, b1,
                                                  (short)0, c1, false, false);
      h8 d0, d1;
#pragma unroll
      for (int r = 0; r < 8; ++r) { d0[r] = (_Float16)c0[r]; d1[r] = (_Float16)c1[r]; }
      *(h8*)(&p_lds[dst0]) = d0;
      *(h8*)(&p_lds[dst1]) = d1;
#endif
    }
  }
  __syncthreads();

  // --- Phase 3: cost[d,h,w] = bias + sum_{i,j,k} P[tap][(d+i-1,h+j-1,w+k-1)] ---
  if (tid < TD * TH * TW) {
    const int wx = tid & 15;
    const int hy = (tid >> 4) & 3;
    const int dz = tid >> 6;
    float acc = conv_b[0];
#pragma unroll
    for (int i = 0; i < 3; ++i)
#pragma unroll
      for (int j = 0; j < 3; ++j)
#pragma unroll
        for (int k = 0; k < 3; ++k) {
          int p = ((dz + i) * PH + (hy + j)) * PW + (wx + k);
          acc += (float)p_lds[(i * 9 + j * 3 + k) * NPOS + p];
        }
    out[((d0 + dz) * Hn + (h0 + hy)) * Wn + (w0 + wx)] = acc;
  }
}

// ---------------------------------------------------------------------------
// In-place softmax over depth (D=64) per (h,w) column.
// ---------------------------------------------------------------------------
__global__ void depthnet_softmax(float* __restrict__ out) {
  __shared__ float red[64];
  const int p = blockIdx.x;     // pixel 0 .. H*W-1
  const int t = threadIdx.x;    // 0 .. 63
  float v = out[t * HWn + p];
  red[t] = v;
  __syncthreads();
  for (int s = 32; s > 0; s >>= 1) {
    if (t < s) red[t] = fmaxf(red[t], red[t + s]);
    __syncthreads();
  }
  float mx = red[0];
  __syncthreads();
  float e = __expf(v - mx);
  red[t] = e;
  __syncthreads();
  for (int s = 32; s > 0; s >>= 1) {
    if (t < s) red[t] += red[t + s];
    __syncthreads();
  }
  out[t * HWn + p] = e / red[0];
}

// ---------------------------------------------------------------------------
extern "C" void kernel_launch(void* const* d_in, const int* in_sizes, int n_in,
                              void* d_out, int out_size, void* d_ws, size_t ws_size,
                              hipStream_t stream) {
  (void)in_sizes; (void)n_in; (void)out_size; (void)ws_size;
  const float* f0     = (const float*)d_in[0];
  const float* f1     = (const float*)d_in[1];
  const float* f2     = (const float*)d_in[2];
  const float* f3     = (const float*)d_in[3];
  const float* f4     = (const float*)d_in[4];
  const float* proj   = (const float*)d_in[5];
  const float* depth  = (const float*)d_in[6];
  const float* conv_w = (const float*)d_in[8];
  const float* conv_b = (const float*)d_in[9];
  float* rt  = (float*)d_ws;       // 4 views x 12 floats
  float* out = (float*)d_out;      // cost then softmax in place

  depthnet_setup<<<1, 1, 0, stream>>>(proj, rt);
  depthnet_costvol<<<10 * 32 * 32, 256, 0, stream>>>(f0, f1, f2, f3, f4, rt,
                                                     depth, conv_w, conv_b, out);
  depthnet_softmax<<<HWn, 64, 0, stream>>>(out);
}
